// GCNBlock_37830071943783
// MI455X (gfx1250) — matrix operand order repaired and checked by reading, not compiled, not run on hardware
//
#include <hip/hip_runtime.h>

typedef __attribute__((ext_vector_type(16))) __bf16 v16bf;
typedef __attribute__((ext_vector_type(8)))  __bf16 v8bf;
typedef __attribute__((ext_vector_type(8)))  float  v8f;

// Shapes
constexpr int cB = 16, cN = 1024, cF = 3, cT = 12, cK = 3, cC = 64;
constexpr int cNFT = cN * cF * cT;       // 36864 per batch in x
constexpr int cFT  = cF * cT;            // 36
constexpr int cCT  = cC * cT;            // 768
constexpr int TOT  = cB * cN * cC * cT;  // 12582912 output elements

__device__ __forceinline__ float sigmoidf_(float z) { return 1.0f / (1.0f + expf(-z)); }

// ---------------- temporal attention ----------------
// tmp1[b,t,f] = sum_n x[b,n,f,t] * U1[n]
__global__ void k_tmp1(const float* __restrict__ x, const float* __restrict__ U1,
                       float* __restrict__ tmp1) {
  __shared__ float red[256];
  int o = blockIdx.x;                   // (b*T+t)*F+f
  int f = o % cF, t = (o / cF) % cT, b = o / (cF * cT);
  float a = 0.f;
  for (int n = threadIdx.x; n < cN; n += 256)
    a += x[((b * cN + n) * cF + f) * cT + t] * U1[n];
  red[threadIdx.x] = a; __syncthreads();
  for (int s = 128; s > 0; s >>= 1) {
    if (threadIdx.x < s) red[threadIdx.x] += red[threadIdx.x + s];
    __syncthreads();
  }
  if (threadIdx.x == 0) tmp1[o] = red[0];
}

// lhs[b,t,j] = sum_f tmp1[b,t,f] * U2[f,j]
__global__ void k_lhs(const float* __restrict__ tmp1, const float* __restrict__ U2,
                      float* __restrict__ lhs) {
  int idx = blockIdx.x * 256 + threadIdx.x;          // B*T*N
  int j = idx % cN, t = (idx / cN) % cT, b = idx / (cN * cT);
  float a = 0.f;
  #pragma unroll
  for (int f = 0; f < cF; ++f) a += tmp1[(b * cT + t) * cF + f] * U2[f * cN + j];
  lhs[idx] = a;
}

// rhsT[b,n,t] = sum_f x[b,n,f,t] * U3[f]
__global__ void k_rhsT(const float* __restrict__ x, const float* __restrict__ U3,
                       float* __restrict__ rhsT) {
  int idx = blockIdx.x * 256 + threadIdx.x;          // B*N*T
  int t = idx % cT, n = (idx / cT) % cN, b = idx / (cT * cN);
  float a = 0.f;
  #pragma unroll
  for (int f = 0; f < cF; ++f) a += x[((b * cN + n) * cF + f) * cT + t] * U3[f];
  rhsT[idx] = a;
}

// sig[b,i,u] = sigmoid(sum_n lhs[b,i,n]*rhsT[b,n,u] + b_e[i,u])
__global__ void k_sigTT(const float* __restrict__ lhs, const float* __restrict__ rhsT,
                        const float* __restrict__ b_e, float* __restrict__ sig) {
  __shared__ float red[256];
  int o = blockIdx.x;                   // (b*T+i)*T+u
  int u = o % cT, i = (o / cT) % cT, b = o / (cT * cT);
  float a = 0.f;
  for (int n = threadIdx.x; n < cN; n += 256)
    a += lhs[(b * cT + i) * cN + n] * rhsT[(b * cN + n) * cT + u];
  red[threadIdx.x] = a; __syncthreads();
  for (int s = 128; s > 0; s >>= 1) {
    if (threadIdx.x < s) red[threadIdx.x] += red[threadIdx.x + s];
    __syncthreads();
  }
  if (threadIdx.x == 0) sig[o] = sigmoidf_(red[0] + b_e[i * cT + u]);
}

// e[b,i,j] = sum_k sig[b,i,k] * V_e[j,k]
__global__ void k_eTT(const float* __restrict__ sig, const float* __restrict__ V_e,
                      float* __restrict__ e) {
  int idx = blockIdx.x * 256 + threadIdx.x;  // B*T*T
  int j = idx % cT, i = (idx / cT) % cT, b = idx / (cT * cT);
  float a = 0.f;
  #pragma unroll
  for (int k = 0; k < cT; ++k) a += sig[(b * cT + i) * cT + k] * V_e[j * cT + k];
  e[idx] = a;
}

// tat[b,:,j] = softmax over axis i of e[b,:,j]
__global__ void k_softT(const float* __restrict__ e, float* __restrict__ tat) {
  int idx = blockIdx.x * 192 + threadIdx.x;  // B*T columns
  int j = idx % cT, b = idx / cT;
  float m = -1e30f;
  float ex[cT];
  for (int i = 0; i < cT; ++i) m = fmaxf(m, e[(b * cT + i) * cT + j]);
  float s = 0.f;
  for (int i = 0; i < cT; ++i) { ex[i] = expf(e[(b * cT + i) * cT + j] - m); s += ex[i]; }
  float inv = 1.f / s;
  for (int i = 0; i < cT; ++i) tat[(b * cT + i) * cT + j] = ex[i] * inv;
}

// x_tat[b,n,f,u] = sum_t x[b,n,f,t] * tat[b,t,u]
__global__ void k_xtat(const float* __restrict__ x, const float* __restrict__ tat,
                       float* __restrict__ xt) {
  int idx = blockIdx.x * 256 + threadIdx.x;  // B*N*F*T
  int u = idx % cT, f = (idx / cT) % cF, n = (idx / cFT) % cN, b = idx / cNFT;
  float a = 0.f;
  #pragma unroll
  for (int t = 0; t < cT; ++t)
    a += x[((b * cN + n) * cF + f) * cT + t] * tat[(b * cT + t) * cT + u];
  xt[idx] = a;
}

// ---------------- spatial attention ----------------
__global__ void k_tmp2(const float* __restrict__ xt, const float* __restrict__ W1,
                       float* __restrict__ tmp2) {
  int idx = blockIdx.x * 256 + threadIdx.x;  // B*N*F
  float a = 0.f;
  #pragma unroll
  for (int t = 0; t < cT; ++t) a += xt[idx * cT + t] * W1[t];
  tmp2[idx] = a;
}

__global__ void k_lhs2(const float* __restrict__ tmp2, const float* __restrict__ W2,
                       float* __restrict__ lhs2) {
  int idx = blockIdx.x * 256 + threadIdx.x;  // B*N*T
  int t = idx % cT, n = (idx / cT) % cN, b = idx / (cT * cN);
  float a = 0.f;
  #pragma unroll
  for (int f = 0; f < cF; ++f) a += tmp2[(b * cN + n) * cF + f] * W2[f * cT + t];
  lhs2[idx] = a;
}

// rhs2[b,t,m] = sum_f x_tat[b,m,f,t] * W3[f]
__global__ void k_rhs2(const float* __restrict__ xt, const float* __restrict__ W3,
                       float* __restrict__ rhs2) {
  int idx = blockIdx.x * 256 + threadIdx.x;  // B*T*N, m fastest
  int m = idx % cN, t = (idx / cN) % cT, b = idx / (cN * cT);
  float a = 0.f;
  #pragma unroll
  for (int f = 0; f < cF; ++f) a += xt[((b * cN + m) * cF + f) * cT + t] * W3[f];
  rhs2[idx] = a;
}

__global__ void k_vsb(const float* __restrict__ Vs, __bf16* __restrict__ vsb) {
  int idx = blockIdx.x * 256 + threadIdx.x;  // N*N
  vsb[idx] = (__bf16)Vs[idx];
}

// sig2[b,n,m] = sigmoid(sum_t lhs2[b,n,t]*rhs2[b,t,m] + b_s[n,m]) -> bf16
__global__ void k_sig2(const float* __restrict__ lhs2, const float* __restrict__ rhs2,
                       const float* __restrict__ b_s, __bf16* __restrict__ sig2b) {
  int idx = blockIdx.x * 256 + threadIdx.x;  // B*N*N, m fastest
  int m = idx % cN, n = (idx / cN) % cN, b = idx / (cN * cN);
  float a = b_s[n * cN + m];
  #pragma unroll
  for (int t = 0; t < cT; ++t)
    a += lhs2[(b * cN + n) * cT + t] * rhs2[(b * cT + t) * cN + m];
  sig2b[idx] = (__bf16)sigmoidf_(a);
}

// ------------- dominant GEMM: s[b,i,j] = sum_k sig2[b,i,k] * V_s[j,k] -------------
// Register-blocked: each wave computes a 32(M) x 64(N) tile = 2x4 WMMA fragments.
// Per 32-wide K step: 4 A b128 loads + 8 B b128 loads feed 8 WMMAs (1.5 loads/wmma);
// A frags reused 4x, B frags 2x from registers.
// A frag (16x32 bf16): lane r=lane&15 is row M=r; lanes>=16 hold K 16..31.
// B frag (32x16 bf16): lane c=lane&15 is col N=c; lanes>=16 hold K 16..31.
// D (16x16 f32): VGPR e holds M = e + 8*(lane>=16), N = lane&15.
__global__ void k_gemm_sat(const __bf16* __restrict__ A, const __bf16* __restrict__ Vs,
                           float* __restrict__ S) {
  const int lane = threadIdx.x & 31;
  const int wave = threadIdx.x >> 5;          // 4 waves; each owns 32 rows
  const int j0 = blockIdx.x * 64;
  const int i0 = blockIdx.y * 128 + wave * 32;
  const int b  = blockIdx.z;
  const int r  = lane & 15;
  const int hi = lane >> 4;
  const __bf16* Ar0 = A + ((size_t)b * cN + (i0 + r)) * cN;   // rows i0..i0+15
  const __bf16* Ar1 = Ar0 + (size_t)16 * cN;                  // rows i0+16..i0+31
  const __bf16* Br0 = Vs + (size_t)(j0 + r) * cN;             // V_s row j == B column j
  const __bf16* Br1 = Br0 + (size_t)16 * cN;
  const __bf16* Br2 = Br0 + (size_t)32 * cN;
  const __bf16* Br3 = Br0 + (size_t)48 * cN;
  v8f acc[2][4] = {};
  for (int kk = 0; kk < cN; kk += 32) {
    v16bf af[2], bfv[4];
    {
      v8bf a0 = *(const v8bf*)(Ar0 + kk + hi * 16);
      v8bf a1 = *(const v8bf*)(Ar0 + kk + hi * 16 + 8);
      v8bf a2 = *(const v8bf*)(Ar1 + kk + hi * 16);
      v8bf a3 = *(const v8bf*)(Ar1 + kk + hi * 16 + 8);
      #pragma unroll
      for (int e = 0; e < 8; ++e) {
        af[0][e] = a0[e]; af[0][e + 8] = a1[e];
        af[1][e] = a2[e]; af[1][e + 8] = a3[e];
      }
    }
    {
      v8bf b0 = *(const v8bf*)(Br0 + kk + hi * 16);
      v8bf b1 = *(const v8bf*)(Br0 + kk + hi * 16 + 8);
      v8bf b2 = *(const v8bf*)(Br1 + kk + hi * 16);
      v8bf b3 = *(const v8bf*)(Br1 + kk + hi * 16 + 8);
      v8bf b4 = *(const v8bf*)(Br2 + kk + hi * 16);
      v8bf b5 = *(const v8bf*)(Br2 + kk + hi * 16 + 8);
      v8bf b6 = *(const v8bf*)(Br3 + kk + hi * 16);
      v8bf b7 = *(const v8bf*)(Br3 + kk + hi * 16 + 8);
      #pragma unroll
      for (int e = 0; e < 8; ++e) {
        bfv[0][e] = b0[e]; bfv[0][e + 8] = b1[e];
        bfv[1][e] = b2[e]; bfv[1][e + 8] = b3[e];
        bfv[2][e] = b4[e]; bfv[2][e + 8] = b5[e];
        bfv[3][e] = b6[e]; bfv[3][e + 8] = b7[e];
      }
    }
    if (kk + 32 < cN) {
      __builtin_prefetch(Ar0 + kk + 32, 0, 0);    // global_prefetch_b8
      __builtin_prefetch(Ar1 + kk + 32, 0, 0);
    }
    #pragma unroll
    for (int mi = 0; mi < 2; ++mi)
      #pragma unroll
      for (int jj = 0; jj < 4; ++jj)
        acc[mi][jj] = __builtin_amdgcn_wmma_f32_16x16x32_bf16(
            /*neg_a=*/false, af[mi], /*neg_b=*/false, bfv[jj],
            /*c_mod=*/(short)0, acc[mi][jj], /*reuse_a=*/false, /*reuse_b=*/false);
  }
  #pragma unroll
  for (int mi = 0; mi < 2; ++mi)
    #pragma unroll
    for (int jj = 0; jj < 4; ++jj) {
      float* Sp = S + ((size_t)b * cN + (i0 + mi * 16 + hi * 8)) * cN + j0 + jj * 16 + r;
      #pragma unroll
      for (int e = 0; e < 8; ++e) Sp[(size_t)e * cN] = acc[mi][jj][e];
    }
}

// column softmax stats over axis i of s[b,i,j]: maxbuf, invbuf = 1/sum(exp)
__global__ void k_colstat(const float* __restrict__ S, float* __restrict__ maxbuf,
                          float* __restrict__ invbuf) {
  int idx = blockIdx.x * 256 + threadIdx.x;   // B*N columns, j fastest
  int j = idx % cN, b = idx / cN;
  const float* base = S + (size_t)b * cN * cN + j;
  float m = -1e30f;
  for (int i = 0; i < cN; ++i) m = fmaxf(m, base[(size_t)i * cN]);
  float s = 0.f;
  for (int i = 0; i < cN; ++i) s += expf(base[(size_t)i * cN] - m);
  maxbuf[idx] = m;
  invbuf[idx] = 1.f / s;
}

// rhs_c[b,k,m,ft] = sum_n cheb[k,n,m] * softmax(s)[b,n,m] * x[b,n,ft]
__global__ void k_rhsc(const float* __restrict__ S, const float* __restrict__ maxbuf,
                       const float* __restrict__ invbuf, const float* __restrict__ cheb,
                       const float* __restrict__ x, float* __restrict__ rhsc) {
  __shared__ float wsh[64];
  const int m = blockIdx.x, k = blockIdx.y, b = blockIdx.z;
  const int tid = threadIdx.x;                // 64 threads
  const float mx  = maxbuf[b * cN + m];
  const float inv = invbuf[b * cN + m];
  float acc = 0.f;
  for (int nb = 0; nb < cN; nb += 64) {
    int n = nb + tid;
    float sat = expf(S[((size_t)b * cN + n) * cN + m] - mx) * inv;
    wsh[tid] = cheb[((size_t)k * cN + n) * cN + m] * sat;
    __syncthreads();
    if (tid < cFT) {
      const float* xp = x + ((size_t)b * cN + nb) * cFT + tid;
      #pragma unroll 8
      for (int nn = 0; nn < 64; ++nn) acc += wsh[nn] * xp[nn * cFT];
    }
    __syncthreads();
  }
  if (tid < cFT) rhsc[(((size_t)b * cK + k) * cN + m) * cFT + tid] = acc;
}

// gcn[b,m,c,t] = relu(sum_{k,f} rhs_c[b,k,m,f,t] * Theta[k,f,c])
__global__ void k_gcn(const float* __restrict__ rhsc, const float* __restrict__ Theta,
                      float* __restrict__ gcn) {
  int idx = blockIdx.x * 256 + threadIdx.x;   // B*N*C*T
  int t = idx % cT, c = (idx / cT) % cC, m = (idx / cCT) % cN, b = idx / (cN * cCT);
  float a = 0.f;
  #pragma unroll
  for (int k = 0; k < cK; ++k)
    #pragma unroll
    for (int f = 0; f < cF; ++f)
      a += rhsc[(((size_t)b * cK + k) * cN + m) * cFT + f * cT + t] *
           Theta[(k * cF + f) * cC + c];
  gcn[idx] = fmaxf(a, 0.f);
}

// y = relu(residual 1x1 conv + temporal (1,3) conv); per-block double partial sums
__global__ void k_y(const float* __restrict__ x, const float* __restrict__ gcn,
                    const float* __restrict__ tc_w, const float* __restrict__ tc_b,
                    const float* __restrict__ rc_w, const float* __restrict__ rc_b,
                    float* __restrict__ out, double* __restrict__ part) {
  __shared__ float s0[256], s1[256];
  int idx = blockIdx.x * 256 + threadIdx.x;   // B*N*C*T
  int t = idx % cT, c = (idx / cT) % cC, n = (idx / cCT) % cN, b = idx / (cN * cCT);
  float z = rc_b[c] + tc_b[c];
  #pragma unroll
  for (int f = 0; f < cF; ++f)
    z += rc_w[c * cF + f] * x[((size_t)b * cN + n) * cFT + f * cT + t];
  const float* g = gcn + ((size_t)b * cN + n) * cCT;
  for (int c2 = 0; c2 < cC; ++c2) {
    const float* gr = g + c2 * cT;
    const float* wt = tc_w + (c * cC + c2) * 3;
    if (t > 0)      z += wt[0] * gr[t - 1];
                    z += wt[1] * gr[t];
    if (t < cT - 1) z += wt[2] * gr[t + 1];
  }
  float y = fmaxf(z, 0.f);
  out[idx] = y;
  s0[threadIdx.x] = y; s1[threadIdx.x] = y * y; __syncthreads();
  for (int s = 128; s > 0; s >>= 1) {
    if (threadIdx.x < s) { s0[threadIdx.x] += s0[threadIdx.x + s]; s1[threadIdx.x] += s1[threadIdx.x + s]; }
    __syncthreads();
  }
  if (threadIdx.x == 0) { part[2 * blockIdx.x] = (double)s0[0]; part[2 * blockIdx.x + 1] = (double)s1[0]; }
}

// deterministic single-block reduction of the 49152 block partials
__global__ void k_red(const double* __restrict__ part, double* __restrict__ red) {
  __shared__ double s0[256], s1[256];
  double a = 0.0, c = 0.0;
  for (int i = threadIdx.x; i < TOT / 256; i += 256) { a += part[2 * i]; c += part[2 * i + 1]; }
  s0[threadIdx.x] = a; s1[threadIdx.x] = c; __syncthreads();
  for (int s = 128; s > 0; s >>= 1) {
    if (threadIdx.x < s) { s0[threadIdx.x] += s0[threadIdx.x + s]; s1[threadIdx.x] += s1[threadIdx.x + s]; }
    __syncthreads();
  }
  if (threadIdx.x == 0) { red[0] = s0[0]; red[1] = s1[0]; }
}

// full-tensor LayerNorm, in place on d_out
__global__ void k_norm(float* __restrict__ out, const double* __restrict__ red,
                       const float* __restrict__ ln_g, const float* __restrict__ ln_b) {
  int idx = blockIdx.x * 256 + threadIdx.x;
  double mu = red[0] / (double)TOT;
  double var = red[1] / (double)TOT - mu * mu;
  float r = rsqrtf((float)var + 1e-5f);
  out[idx] = (out[idx] - (float)mu) * r * ln_g[idx] + ln_b[idx];
}

extern "C" void kernel_launch(void* const* d_in, const int* in_sizes, int n_in,
                              void* d_out, int out_size, void* d_ws, size_t ws_size,
                              hipStream_t stream) {
  const float* x    = (const float*)d_in[0];
  const float* U1   = (const float*)d_in[1];
  const float* U2   = (const float*)d_in[2];
  const float* U3   = (const float*)d_in[3];
  const float* b_e  = (const float*)d_in[4];
  const float* V_e  = (const float*)d_in[5];
  const float* W1   = (const float*)d_in[6];
  const float* W2   = (const float*)d_in[7];
  const float* W3   = (const float*)d_in[8];
  const float* b_s  = (const float*)d_in[9];
  const float* V_s  = (const float*)d_in[10];
  const float* cheb = (const float*)d_in[11];
  const float* Theta= (const float*)d_in[12];
  const float* tc_w = (const float*)d_in[13];
  const float* tc_b = (const float*)d_in[14];
  const float* rc_w = (const float*)d_in[15];
  const float* rc_b = (const float*)d_in[16];
  const float* ln_g = (const float*)d_in[17];
  const float* ln_b = (const float*)d_in[18];
  float* out = (float*)d_out;

  char* base = (char*)d_ws;
  size_t off = 0;
  auto alloc = [&](size_t bytes) -> void* {
    void* p = base + off;
    off = (off + bytes + 255) & ~(size_t)255;
    return p;
  };
  float*  tmp1   = (float*)alloc(sizeof(float) * cB * cT * cF);
  float*  lhs    = (float*)alloc(sizeof(float) * cB * cT * cN);
  float*  rhsT   = (float*)alloc(sizeof(float) * cB * cN * cT);
  float*  sig    = (float*)alloc(sizeof(float) * cB * cT * cT);
  float*  e      = (float*)alloc(sizeof(float) * cB * cT * cT);
  float*  tat    = (float*)alloc(sizeof(float) * cB * cT * cT);
  float*  xt     = (float*)alloc(sizeof(float) * cB * cNFT);
  float*  tmp2   = (float*)alloc(sizeof(float) * cB * cN * cF);
  float*  lhs2   = (float*)alloc(sizeof(float) * cB * cN * cT);
  float*  rhs2   = (float*)alloc(sizeof(float) * cB * cT * cN);
  __bf16* sig2b  = (__bf16*)alloc(sizeof(__bf16) * (size_t)cB * cN * cN);
  __bf16* vsb    = (__bf16*)alloc(sizeof(__bf16) * (size_t)cN * cN);
  float*  S      = (float*)alloc(sizeof(float) * (size_t)cB * cN * cN);
  float*  maxbuf = (float*)alloc(sizeof(float) * cB * cN);
  float*  invbuf = (float*)alloc(sizeof(float) * cB * cN);
  float*  rhsc   = (float*)alloc(sizeof(float) * (size_t)cB * cK * cN * cFT);
  float*  gcn    = (float*)alloc(sizeof(float) * (size_t)TOT);
  double* part   = (double*)alloc(sizeof(double) * 2 * (TOT / 256));
  double* red    = (double*)alloc(sizeof(double) * 2);
  (void)ws_size; (void)n_in; (void)in_sizes; (void)out_size;

  // temporal attention
  k_tmp1 <<<cB * cT * cF, 256, 0, stream>>>(x, U1, tmp1);
  k_lhs  <<<cB * cT * cN / 256, 256, 0, stream>>>(tmp1, U2, lhs);
  k_rhsT <<<cB * cN * cT / 256, 256, 0, stream>>>(x, U3, rhsT);
  k_sigTT<<<cB * cT * cT, 256, 0, stream>>>(lhs, rhsT, b_e, sig);
  k_eTT  <<<cB * cT * cT / 256, 256, 0, stream>>>(sig, V_e, e);
  k_softT<<<1, cB * cT, 0, stream>>>(e, tat);
  k_xtat <<<cB * cNFT / 256, 256, 0, stream>>>(x, tat, xt);

  // spatial attention
  k_tmp2 <<<cB * cN * cF / 256, 256, 0, stream>>>(xt, W1, tmp2);
  k_lhs2 <<<cB * cN * cT / 256, 256, 0, stream>>>(tmp2, W2, lhs2);
  k_rhs2 <<<cB * cT * cN / 256, 256, 0, stream>>>(xt, W3, rhs2);
  k_vsb  <<<cN * cN / 256, 256, 0, stream>>>(V_s, vsb);
  k_sig2 <<<(int)((size_t)cB * cN * cN / 256), 256, 0, stream>>>(lhs2, rhs2, b_s, sig2b);

  // dominant batched GEMM via register-blocked bf16 WMMA (16 batches of 1024^3)
  k_gemm_sat<<<dim3(cN / 64, cN / 128, cB), 128, 0, stream>>>(sig2b, vsb, S);

  // spatial softmax stats + cheb conv + projection
  k_colstat<<<cB * cN / 256, 256, 0, stream>>>(S, maxbuf, invbuf);
  k_rhsc<<<dim3(cN, cK, cB), 64, 0, stream>>>(S, maxbuf, invbuf, cheb, x, rhsc);
  k_gcn <<<TOT / 256, 256, 0, stream>>>(rhsc, Theta, gcn);

  // temporal conv + residual conv + relu, with partial sums for LayerNorm
  k_y   <<<TOT / 256, 256, 0, stream>>>(x, gcn, tc_w, tc_b, rc_w, rc_b, out, part);
  k_red <<<1, 256, 0, stream>>>(part, red);
  k_norm<<<TOT / 256, 256, 0, stream>>>(out, red, ln_g, ln_b);
}
